// VQ_CVAE_62689342652551
// MI455X (gfx1250) — compile-verified, matrix operand order checked by reading, not run on hardware
//
#include <hip/hip_runtime.h>

// ---------------------------------------------------------------------------
// CDNA5 / gfx1250 VQ-CVAE forward.
// wave32; WMMA bf16 16x16x32 for decoder convs (bf16 NHWC activations,
// B-operand pre-packed to lane layout), WMMA f32 16x16x4 for the VQ GEMM.
// ---------------------------------------------------------------------------

typedef __attribute__((ext_vector_type(16))) __bf16 v16bf;
typedef __attribute__((ext_vector_type(8)))  __bf16 v8bf;
typedef __attribute__((ext_vector_type(8)))  float  v8f;
typedef __attribute__((ext_vector_type(2)))  float  v2f;

#define BN_EPS_RSQRT 0.99999500003749978f  /* rsqrt(1+1e-5) */

__device__ __forceinline__ int koff16(int j, int half) {
  // 16-bit A/B operand K-offset for element j of the per-lane v16bf
  return (j < 8) ? (half * 8 + j) : (16 + half * 8 + (j - 8));
}

__device__ __forceinline__ int iclamp(int v, int lo, int hi) {
  return v < lo ? lo : (v > hi ? hi : v);
}

// ===========================================================================
// AdderNet L1 "conv":  y[b,co,l] = -sum_p |patch[b,p,l] - w[co,p]|
// EPI 0: out = relu(y*s + bb)         EPI 1: out = (y + res)*s + bb
// ===========================================================================
template<int CIN, int KS, int STRIDE, int PAD, bool PRERELU, int EPI>
__global__ void adder_k(const float* __restrict__ in, const float* __restrict__ w,
                        const float* __restrict__ gam, const float* __restrict__ bet,
                        const float* __restrict__ res, float* __restrict__ out,
                        int H, int W, int Ho, int Wo, int COUT)
{
  constexpr int P = CIN * KS * KS;
  __shared__ float wl[P];
  int b  = blockIdx.x / COUT;
  int co = blockIdx.x % COUT;
  for (int i = threadIdx.x; i < P; i += blockDim.x) wl[i] = w[(size_t)co * P + i];
  __syncthreads();

  float s = gam[co] * BN_EPS_RSQRT;
  float bb = bet[co];
  const float* inb = in + (size_t)b * CIN * H * W;
  int L = Ho * Wo;
  for (int l = threadIdx.x; l < L; l += blockDim.x) {
    int ho = l / Wo, wo = l % Wo;
    int hi0 = ho * STRIDE - PAD, wi0 = wo * STRIDE - PAD;
    float acc = 0.0f;
    for (int c = 0; c < CIN; ++c) {
      const float* ic = inb + (size_t)c * H * W;
      const float* wc = &wl[c * KS * KS];
#pragma unroll
      for (int kh = 0; kh < KS; ++kh) {
        int hh = hi0 + kh;
#pragma unroll
        for (int kw = 0; kw < KS; ++kw) {
          int ww = wi0 + kw;
          float v = 0.0f;
          if (hh >= 0 && hh < H && ww >= 0 && ww < W) {
            v = ic[hh * W + ww];
            if (PRERELU) v = fmaxf(v, 0.0f);
          }
          acc += fabsf(v - wc[kh * KS + kw]);
        }
      }
    }
    float y = -acc;
    size_t oidx = (((size_t)b * COUT + co) * Ho + ho) * Wo + wo;
    float o;
    if (EPI == 0) o = fmaxf(y * s + bb, 0.0f);
    else          o = (y + res[oidx]) * s + bb;
    out[oidx] = o;
  }
}

// ===========================================================================
// VQ: e2[k] = sum_c E[c,k]^2
// ===========================================================================
__global__ void e2_k(const float* __restrict__ E, float* __restrict__ e2)
{
  int k = blockIdx.x * blockDim.x + threadIdx.x;
  if (k >= 512) return;
  float s = 0.0f;
  for (int c = 0; c < 128; ++c) { float t = E[c * 512 + k]; s += t * t; }
  e2[k] = s;
}

// ===========================================================================
// VQ nearest-embed: scores = -2 * (flat @ E) + e2, argmin over 512 codes.
// f32 WMMA 16x16x4; one wave = 16-row strip x 512 codes; blockDim = 128.
// B loads use one base pointer per code column + immediate offsets (no
// per-step address registers -> no scratch spills).
// ===========================================================================
__global__ void vq_k(const float* __restrict__ ze, const float* __restrict__ E,
                     const float* __restrict__ e2, float* __restrict__ am_out,
                     int* __restrict__ idx_out)
{
  __shared__ float Al[4][16 * 128];
  __shared__ float e2l[512];
  int tid  = threadIdx.x;
  int wave = tid >> 5;
  int lane = tid & 31;
  int half = lane >> 4;
  int lm   = lane & 15;
  for (int i = tid; i < 512; i += 128) e2l[i] = e2[i];

  int strip = blockIdx.x * 4 + wave;          // 512 strips of 16 rows
  for (int i = lane; i < 16 * 128; i += 32) {
    int m = i >> 7, c = i & 127;
    int n = strip * 16 + m;
    int b = n >> 10, hw = n & 1023;
    Al[wave][m * 128 + c] = ze[(((size_t)b * 128 + c) << 10) + hw];
  }
  __syncthreads();
  const float* Arow = &Al[wave][0] + lm * 128 + half * 2;

  float bestv[8];
  int   besti[8];
#pragma unroll
  for (int v = 0; v < 8; ++v) { bestv[v] = 3.4e38f; besti[v] = 0; }

  for (int nt = 0; nt < 32; ++nt) {
    int code = nt * 16 + lm;
    const float* Eb = E + code + half * 2 * 512; // rows (half*2, half*2+1)
    v8f acc;
#pragma unroll
    for (int q = 0; q < 8; ++q) acc[q] = 0.0f;
#pragma unroll 8
    for (int k4 = 0; k4 < 32; ++k4) {            // kb = k4*4
      v2f a, bm;
      a.x  = Arow[k4 * 4];
      a.y  = Arow[k4 * 4 + 1];
      bm.x = Eb[(size_t)k4 * 2048];              // (k4*4)*512
      bm.y = Eb[(size_t)k4 * 2048 + 512];
      acc = __builtin_amdgcn_wmma_f32_16x16x4_f32(false, a, false, bm,
                                                  (short)0, acc, false, false);
    }
#pragma unroll
    for (int v = 0; v < 8; ++v) {
      float sc = -2.0f * acc[v] + e2l[code];
      if (sc < bestv[v]) { bestv[v] = sc; besti[v] = code; }
    }
  }
#pragma unroll
  for (int off = 1; off < 16; off <<= 1) {
#pragma unroll
    for (int v = 0; v < 8; ++v) {
      float ov = __shfl_xor(bestv[v], off, 16);
      int   oi = __shfl_xor(besti[v], off, 16);
      if (ov < bestv[v] || (ov == bestv[v] && oi < besti[v])) {
        bestv[v] = ov; besti[v] = oi;
      }
    }
  }
  if (lm == 0) {
#pragma unroll
    for (int v = 0; v < 8; ++v) {
      int n = strip * 16 + v + 8 * half;
      am_out[n]  = (float)besti[v];
      idx_out[n] = besti[v];
    }
  }
}

// emb[b,c,h,w] = E[c, idx[n]]  (f32 NCHW output) and bf16 NHWC relu'd copy
// (decoder conv1 consumes relu(z_q); relu commutes with bf16 rounding).
__global__ void gather_k(const int* __restrict__ idx, const float* __restrict__ E,
                         float* __restrict__ emb, __bf16* __restrict__ emb16)
{
  size_t i = (size_t)blockIdx.x * blockDim.x + threadIdx.x;   // NCHW flat, 1M
  int w = i & 31, h = (i >> 5) & 31, c = (i >> 10) & 127, b = (int)(i >> 17);
  int n = (b << 10) + (h << 5) + w;
  float v = E[(size_t)c * 512 + idx[n]];
  emb[i] = v;
  emb16[(size_t)n * 128 + c] = (__bf16)fmaxf(v, 0.0f);
}

// ===========================================================================
// Weight pre-pack into WMMA-B lane layout (bf16):
//   out[(((sh*(CIN/32)+ks)*NT + nt)*32 + lane)*16 + j] = w[co, c, kh, kw]
//   co = nt*16 + lane%16 ; c = ks*32 + koff16(j, lane/16)
// ===========================================================================
template<int KS, int NT>
__global__ void pack_k(const float* __restrict__ w, __bf16* __restrict__ out, int COUT)
{
  const int CIN = 128;
  int total = KS * KS * (CIN / 32) * NT * 32 * 16;
  int i = blockIdx.x * blockDim.x + threadIdx.x;
  if (i >= total) return;
  int j    = i & 15;
  int lane = (i >> 4) & 31;
  int q    = i >> 9;
  int nt   = q % NT;  q /= NT;
  int ks   = q % (CIN / 32);
  int sh   = q / (CIN / 32);
  int c  = ks * 32 + koff16(j, lane >> 4);
  int co = nt * 16 + (lane & 15);
  float v = 0.0f;
  if (co < COUT)
    v = w[(((size_t)co * CIN + c) * KS + sh / KS) * KS + (sh % KS)];
  out[i] = (__bf16)v;
}

// ===========================================================================
// Implicit-GEMM conv (CIN=128) via WMMA bf16 16x16x32.
// Input activations: bf16 NHWC  ->  A half-rows are two aligned b128 loads.
// conv3x3 = 9 shifted 1x1 GEMMs. One wave: 16 w-positions x NT*16 couts.
// EPI 0: relu(acc+bias)              EPI 1: ((acc+bias)+res)*s+bb
// EPI 2: relu((acc+bias)*s+bb)       EPI 3: tanh(acc+bias), mask co<COUT
// EPI 4: (acc+bias)+res
// OUT32: write f32 NCHW; OUT16: write bf16 NHWC (RELU16: relu'd copy).
// ===========================================================================
template<int KS, int NT, int EPI, bool OUT32, bool OUT16, bool RELU16>
__global__ void conv_k(const __bf16* __restrict__ in, const __bf16* __restrict__ wp,
                       const float* __restrict__ bias, const float* __restrict__ res,
                       const float* __restrict__ gam, const float* __restrict__ bet,
                       float* __restrict__ out, __bf16* __restrict__ out16,
                       int H, int W, int COUT)
{
  const int CIN = 128;
  int tid  = threadIdx.x;
  int wave = tid >> 5;
  int lane = tid & 31;
  int half = lane >> 4;
  int lm   = lane & 15;

  int mt    = blockIdx.x * 8 + wave;
  int per_b = (H * W) / 16;
  int b = mt / per_b;
  int t = mt % per_b;
  int w0 = (t * 16) % W;
  int h  = (t * 16) / W;

  v8f acc[NT];
#pragma unroll
  for (int nt = 0; nt < NT; ++nt)
#pragma unroll
    for (int q = 0; q < 8; ++q) acc[nt][q] = 0.0f;

  const __bf16* inb = in + (size_t)b * H * W * CIN;

#pragma unroll
  for (int sh = 0; sh < KS * KS; ++sh) {
    int dh = sh / KS - KS / 2;
    int dw = sh % KS - KS / 2;
    int hh = h + dh;
    int ww = w0 + lm + dw;        // this lane's A-row spatial column
    bool ok = (hh >= 0 && hh < H && ww >= 0 && ww < W);
    const __bf16* pix = inb + ((size_t)(hh * W + ww)) * CIN + half * 8;
#pragma unroll
    for (int ks = 0; ks < CIN / 32; ++ks) {
      int kb = ks * 32;
      v16bf a;
      if (ok) {
        v8bf lo = *(const v8bf*)(pix + kb);
        v8bf hi = *(const v8bf*)(pix + kb + 16);
#pragma unroll
        for (int j = 0; j < 8; ++j) { a[j] = lo[j]; a[j + 8] = hi[j]; }
      } else {
#pragma unroll
        for (int j = 0; j < 16; ++j) a[j] = (__bf16)0.0f;
      }
#pragma unroll
      for (int nt = 0; nt < NT; ++nt) {
        const __bf16* bp =
            wp + ((((size_t)sh * (CIN / 32) + ks) * NT + nt) * 32 + lane) * 16;
        v16bf bm = *(const v16bf*)bp;
        acc[nt] = __builtin_amdgcn_wmma_f32_16x16x32_bf16(false, a, false, bm,
                                                          (short)0, acc[nt],
                                                          false, false);
      }
    }
  }

  // Epilogue: lane holds col co = nt*16+lm, rows m = v + 8*half (w = w0+m)
#pragma unroll
  for (int nt = 0; nt < NT; ++nt) {
    int co = nt * 16 + lm;
    float bs = (co < COUT) ? bias[co] : 0.0f;
    float s = 1.0f, bb = 0.0f;
    if (EPI == 1 || EPI == 2) { s = gam[co] * BN_EPS_RSQRT; bb = bet[co]; }
#pragma unroll
    for (int v = 0; v < 8; ++v) {
      int wo = w0 + v + 8 * half;
      float y = acc[nt][v] + bs;
      if (EPI == 0)      y = fmaxf(y, 0.0f);
      else if (EPI == 1) y = (y + res[(((size_t)b * 128 + co) * H + h) * W + wo]) * s + bb;
      else if (EPI == 2) y = fmaxf(y * s + bb, 0.0f);
      else if (EPI == 3) y = tanhf(y);
      else if (EPI == 4) y = y + res[(((size_t)b * 128 + co) * H + h) * W + wo];
      if (OUT32 && co < COUT)
        out[(((size_t)b * COUT + co) * H + h) * W + wo] = y;
      if (OUT16) {
        float t16 = RELU16 ? fmaxf(y, 0.0f) : y;
        out16[(((size_t)b * H + h) * W + wo) * 128 + co] = (__bf16)t16;
      }
    }
  }
}

// ===========================================================================
// Bilinear 2x upsample (half-pixel, edge clamp), f32 NCHW -> bf16 NHWC.
// ===========================================================================
__global__ void up2_k(const float* __restrict__ in, __bf16* __restrict__ out16,
                      int C, int H, int W)
{
  int Ho = 2 * H, Wo = 2 * W;
  size_t i = (size_t)blockIdx.x * blockDim.x + threadIdx.x;  // NCHW flat
  size_t total = (size_t)8 * C * Ho * Wo;
  if (i >= total) return;
  int wo = (int)(i % Wo);
  int ho = (int)((i / Wo) % Ho);
  size_t bc = i / ((size_t)Wo * Ho);
  int c = (int)(bc % C);
  int b = (int)(bc / C);
  float fh = (ho + 0.5f) * 0.5f - 0.5f;
  float fw = (wo + 0.5f) * 0.5f - 0.5f;
  int h0 = (int)floorf(fh); float th = fh - (float)h0;
  int w0 = (int)floorf(fw); float tw = fw - (float)w0;
  int h0c = iclamp(h0, 0, H - 1), h1c = iclamp(h0 + 1, 0, H - 1);
  int w0c = iclamp(w0, 0, W - 1), w1c = iclamp(w0 + 1, 0, W - 1);
  const float* p = in + bc * (size_t)H * W;
  float v00 = p[h0c * W + w0c], v01 = p[h0c * W + w1c];
  float v10 = p[h1c * W + w0c], v11 = p[h1c * W + w1c];
  float v = (1.0f - th) * ((1.0f - tw) * v00 + tw * v01)
          + th          * ((1.0f - tw) * v10 + tw * v11);
  out16[(((size_t)b * Ho + ho) * Wo + wo) * (size_t)C + c] = (__bf16)v;
}

// ===========================================================================
// Host launcher
// ===========================================================================
extern "C" void kernel_launch(void* const* d_in, const int* in_sizes, int n_in,
                              void* d_out, int out_size, void* d_ws, size_t ws_size,
                              hipStream_t stream)
{
  (void)in_sizes; (void)n_in; (void)out_size; (void)ws_size;
  const float* x      = (const float*)d_in[0];
  const float* we1    = (const float*)d_in[1];
  const float* we2    = (const float*)d_in[2];
  const float* ae1_w1 = (const float*)d_in[3];
  const float* ae1_w2 = (const float*)d_in[4];
  const float* ae2_w1 = (const float*)d_in[5];
  const float* ae2_w2 = (const float*)d_in[6];
  const float* bng    = (const float*)d_in[7];   // [8,128]
  const float* bnb    = (const float*)d_in[8];
  const float* E      = (const float*)d_in[9];   // [128,512]
  const float* rd1_w1 = (const float*)d_in[10];
  const float* rd1_b1 = (const float*)d_in[11];
  const float* rd1_w2 = (const float*)d_in[12];
  const float* rd1_b2 = (const float*)d_in[13];
  const float* rd2_w1 = (const float*)d_in[14];
  const float* rd2_b1 = (const float*)d_in[15];
  const float* rd2_w2 = (const float*)d_in[16];
  const float* rd2_b2 = (const float*)d_in[17];
  const float* cd1_w  = (const float*)d_in[18];
  const float* cd1_b  = (const float*)d_in[19];
  const float* cd2_w  = (const float*)d_in[20];
  const float* cd2_b  = (const float*)d_in[21];

  const float* g0 = bng + 0 * 128, *b0 = bnb + 0 * 128;
  const float* g1 = bng + 1 * 128, *b1 = bnb + 1 * 128;
  const float* g2 = bng + 2 * 128, *b2 = bnb + 2 * 128;
  const float* g3 = bng + 3 * 128, *b3 = bnb + 3 * 128;
  const float* g4 = bng + 4 * 128, *b4 = bnb + 4 * 128;
  const float* g5 = bng + 5 * 128, *b5 = bnb + 5 * 128;
  const float* g6 = bng + 6 * 128, *b6 = bnb + 6 * 128;
  const float* g7 = bng + 7 * 128, *b7 = bnb + 7 * 128;

  // d_out regions: y | z_e | emb | argmin(float)
  float* outf = (float*)d_out;
  float* y_o   = outf;
  float* ze_o  = outf + 393216;
  float* emb_o = outf + 1441792;
  float* am_o  = outf + 2490368;

  // workspace layout (256B aligned)
  char* base = (char*)d_ws;
  size_t off = 0;
  auto alloc = [&](size_t bytes) { char* p = base + off; off = (off + bytes + 255) & ~(size_t)255; return p; };
  float*  buf1  = (float*) alloc(16777216);  // 8x128x64x64 f32 (encoder)
  float*  buf2  = (float*) alloc(4194304);   // 8x128x32x32 f32
  float*  buf3  = (float*) alloc(4194304);
  float*  buf4  = (float*) alloc(4194304);
  float*  buf5  = (float*) alloc(16777216);  // 8x128x64x64 f32 (cd1 out)
  float*  e2f   = (float*) alloc(2048);
  int*    idxw  = (int*)   alloc(32768);
  __bf16* p1    = (__bf16*)alloc(294912);
  __bf16* p2    = (__bf16*)alloc(32768);
  __bf16* p3    = (__bf16*)alloc(294912);
  __bf16* p4    = (__bf16*)alloc(32768);
  __bf16* p5    = (__bf16*)alloc(294912);
  __bf16* p6    = (__bf16*)alloc(36864);
  __bf16* emb16 = (__bf16*)alloc(2097152);   // 8x32x32x128 bf16 NHWC (relu'd z_q)
  __bf16* h16a  = (__bf16*)alloc(2097152);
  __bf16* h16b  = (__bf16*)alloc(2097152);
  __bf16* u16a  = (__bf16*)alloc(8388608);   // 8x64x64x128
  __bf16* u16b  = (__bf16*)alloc(33554432);  // 8x128x128x128

  // --- weight pre-pack + code norms (input-only dependence) ---
  pack_k<3, 8><<<576, 256, 0, stream>>>(rd1_w1, p1, 128);
  pack_k<1, 8><<<64,  256, 0, stream>>>(rd1_w2, p2, 128);
  pack_k<3, 8><<<576, 256, 0, stream>>>(rd2_w1, p3, 128);
  pack_k<1, 8><<<64,  256, 0, stream>>>(rd2_w2, p4, 128);
  pack_k<3, 8><<<576, 256, 0, stream>>>(cd1_w,  p5, 128);
  pack_k<3, 1><<<72,  256, 0, stream>>>(cd2_w,  p6, 3);
  e2_k<<<2, 256, 0, stream>>>(E, e2f);

  // --- encoder (AdderNet, VALU L1 distances) ---
  adder_k<3,   4, 2, 1, false, 0><<<1024, 256, 0, stream>>>(x,    we1,    g0, b0, nullptr, buf1, 128, 128, 64, 64, 128);
  adder_k<128, 4, 2, 1, false, 0><<<1024, 256, 0, stream>>>(buf1, we2,    g1, b1, nullptr, buf2, 64, 64, 32, 32, 128);
  adder_k<128, 3, 1, 1, true,  0><<<1024, 256, 0, stream>>>(buf2, ae1_w1, g2, b2, nullptr, buf3, 32, 32, 32, 32, 128);
  adder_k<128, 1, 1, 0, false, 1><<<1024, 256, 0, stream>>>(buf3, ae1_w2, g3, b3, buf2,    buf4, 32, 32, 32, 32, 128);
  adder_k<128, 3, 1, 1, true,  0><<<1024, 256, 0, stream>>>(buf4, ae2_w1, g4, b4, nullptr, buf3, 32, 32, 32, 32, 128);
  adder_k<128, 1, 1, 0, false, 1><<<1024, 256, 0, stream>>>(buf3, ae2_w2, g5, b5, buf4,    ze_o, 32, 32, 32, 32, 128);

  // --- VQ ---
  vq_k<<<128, 128, 0, stream>>>(ze_o, E, e2f, am_o, idxw);
  gather_k<<<4096, 256, 0, stream>>>(idxw, E, emb_o, emb16);   // emb == z_q

  // --- decoder (WMMA bf16 convs; activations bf16 NHWC) ---
  conv_k<3, 8, 0, false, true,  false><<<64,   256, 0, stream>>>(emb16, p1, rd1_b1, nullptr, nullptr, nullptr, nullptr, h16a, 32, 32, 128);
  conv_k<1, 8, 1, true,  true,  true ><<<64,   256, 0, stream>>>(h16a,  p2, rd1_b2, emb_o,   g6,      b6,      buf3,    h16b, 32, 32, 128);
  conv_k<3, 8, 0, false, true,  false><<<64,   256, 0, stream>>>(h16b,  p3, rd2_b1, nullptr, nullptr, nullptr, nullptr, h16a, 32, 32, 128);
  conv_k<1, 8, 4, true,  false, false><<<64,   256, 0, stream>>>(h16a,  p4, rd2_b2, buf3,    nullptr, nullptr, buf4,    nullptr, 32, 32, 128);
  up2_k<<<16384, 256, 0, stream>>>(buf4, u16a, 128, 32, 32);
  conv_k<3, 8, 2, true,  false, false><<<256,  256, 0, stream>>>(u16a,  p5, cd1_b,  nullptr, g7,      b7,      buf5,    nullptr, 64, 64, 128);
  up2_k<<<65536, 256, 0, stream>>>(buf5, u16b, 128, 64, 64);
  conv_k<3, 1, 3, true,  false, false><<<1024, 256, 0, stream>>>(u16b,  p6, cd2_b,  nullptr, nullptr, nullptr, y_o,     nullptr, 128, 128, 3);
}
